// Learnable_32650341384420
// MI455X (gfx1250) — matrix operand order repaired and checked by reading, not compiled, not run on hardware
//
#include <hip/hip_runtime.h>
#include <hip/hip_bf16.h>
#include <math.h>

typedef __attribute__((ext_vector_type(16))) _Float16 v16h;
typedef __attribute__((ext_vector_type(8)))  _Float16 v8h;
typedef __attribute__((ext_vector_type(8)))  float    v8f;

#define B_DIM 4
#define H_DIM 1024
#define L_DIM 8192
#define KD 64
#define KLEN 512
#define KREV_PITCH 576      // padded reversed-filter length per h (36*16)
#define CONV_TILE 2048      // conv outputs per workgroup (8 waves * 256)
#define NSTEPS 17           // K=32 blocks covering s in [-512, 31]
#define ULDS 2592           // u staging: [L0-512, L0+2063] = 2576, padded

#define GM 128
#define GN 128
#define GK 32
#define GKSTEPS (H_DIM / GK)
#define KPITCH 40           // LDS k-pitch in halfs (80B rows, 16B aligned)

#if __has_builtin(__builtin_amdgcn_global_load_async_to_lds_b128) && \
    __has_builtin(__builtin_amdgcn_s_wait_asynccnt)
#define HAS_ASYNC_LDS 1
#else
#define HAS_ASYNC_LDS 0
#endif

static __device__ __forceinline__ v16h cat16(v8h lo, v8h hi) {
  return __builtin_shufflevector(lo, hi, 0,1,2,3,4,5,6,7,8,9,10,11,12,13,14,15);
}

#if HAS_ASYNC_LDS
// Builtin expects: (v4i addrspace(1)*, v4i addrspace(3)*, imm offset, imm cpol)
// where v4i = int __attribute__((vector_size(16))).
typedef int v4i_vs __attribute__((vector_size(4 * sizeof(int))));
typedef __attribute__((address_space(1))) v4i_vs as1_v4i;
typedef __attribute__((address_space(3))) v4i_vs as3_v4i;

// src: global f16 ptr, dst: shared f16 ptr. Flat shared addr low 32 bits are
// the LDS byte offset (ISA 10.2 aperture mapping).
static __device__ __forceinline__ void async_copy16B(const _Float16* src,
                                                     _Float16* dst) {
  as1_v4i* gp = (as1_v4i*)(uintptr_t)src;
  as3_v4i* lp = (as3_v4i*)(unsigned int)(uintptr_t)dst;
  __builtin_amdgcn_global_load_async_to_lds_b128(gp, lp, 0, 0);
}
#endif

// ---------------------------------------------------------------------------
// Stage 1: build per-h normalized filter, store REVERSED + zero-padded (f16):
//   krev[p] = k_norm[527 - p] for p in [16,527], else 0.   (length 576)
// ---------------------------------------------------------------------------
__global__ __launch_bounds__(256)
void build_filter_kernel(const float* __restrict__ k0, const float* __restrict__ k1,
                         const float* __restrict__ k2, const float* __restrict__ k3,
                         _Float16* __restrict__ krev_g) {
  __shared__ float ksum[KLEN];
  __shared__ float red[256];
  const int h = blockIdx.x;
  const int tid = threadIdx.x;
  const float* kins[4] = {k0 + h * KD, k1 + h * KD, k2 + h * KD, k3 + h * KD};

  float ss = 0.f;
  for (int j = tid; j < KLEN; j += 256) {
    float val = 0.f;
    #pragma unroll
    for (int i = 0; i < 4; ++i) {
      const int len = KD << i;
      if (j < len) {
        // F.interpolate(mode='linear', align_corners=False)
        const float scale = (float)(1 << i);
        float coord = ((float)j + 0.5f) / scale - 0.5f;
        coord = fminf(fmaxf(coord, 0.f), 63.f);
        int lo = (int)floorf(coord);
        int hi = min(lo + 1, 63);
        float w = coord - (float)lo;
        float v = kins[i][lo] * (1.f - w) + kins[i][hi] * w;
        val += v * (float)(8 >> i);   // 2^(NS-1-i)
      }
    }
    ksum[j] = val;
    ss += val * val;
  }
  red[tid] = ss;
  __syncthreads();
  for (int off = 128; off > 0; off >>= 1) {
    if (tid < off) red[tid] += red[tid + off];
    __syncthreads();
  }
  const float invn = 1.0f / sqrtf(red[0] + 1e-30f);
  for (int p = tid; p < KREV_PITCH; p += 256) {
    float v = (p >= 16 && p < 528) ? ksum[527 - p] * invn : 0.f;
    krev_g[h * KREV_PITCH + p] = (_Float16)v;
  }
}

// ---------------------------------------------------------------------------
// Stage 1b: W (1024x1024 fp32) -> f16
// ---------------------------------------------------------------------------
__global__ void convert_w_kernel(const float* __restrict__ W, _Float16* __restrict__ W16) {
  int i = blockIdx.x * blockDim.x + threadIdx.x;
  if (i < H_DIM * H_DIM) W16[i] = (_Float16)W[i];
}

// ---------------------------------------------------------------------------
// Stage 2: 512-tap FIR as Toeplitz WMMA + D*u + exact GELU -> f16 buffer G.
// y[n] = sum_s u[n0+s] * k[j-s]:  A[i][s]=u (contiguous), B[s][j]=k[j-s].
// Grid: (B*H, L/2048), 256 threads (8 waves; 256 outputs per wave tile).
// ---------------------------------------------------------------------------
__global__ __launch_bounds__(256)
void conv_gelu_kernel(const float* __restrict__ u, const float* __restrict__ Dcoef,
                      const _Float16* __restrict__ krev_g, _Float16* __restrict__ G) {
  __shared__ _Float16 krev_lds[KREV_PITCH];
  __shared__ _Float16 krevs[16 * KREV_PITCH];   // 16 shifted copies: aligned B frags
  __shared__ _Float16 u_lds[ULDS];

  const int tid = threadIdx.x;
  const int bh = blockIdx.x;
  const int b = bh >> 10, h = bh & 1023;
  const int L0 = blockIdx.y * CONV_TILE;
  const size_t ubase = (size_t)(b * H_DIM + h) * L_DIM;

  for (int p = tid; p < KREV_PITCH; p += 256)
    krev_lds[p] = krev_g[h * KREV_PITCH + p];
  for (int s = tid; s < 2576; s += 256) {
    int gi = L0 - 512 + s;
    float v = (gi >= 0 && gi < L_DIM) ? u[ubase + gi] : 0.f;
    u_lds[s] = (_Float16)v;
  }
  __syncthreads();
  // krevs[j][q] = krev[q - j - 1]; lane with column j reads two 8-half,
  // 16B-aligned runs at q0 = 16 + 32t + 8*grp and q0 + 16.
  for (int idx = tid; idx < 16 * KREV_PITCH; idx += 256) {
    int j = idx / KREV_PITCH;
    int q = idx - j * KREV_PITCH;
    int p = q - j - 1;
    krevs[idx] = (p >= 0 && p < KREV_PITCH) ? krev_lds[p] : (_Float16)0.f;
  }
  __syncthreads();

  const int wid = tid >> 5, lane = tid & 31;
  const int col = lane & 15, grp = lane >> 4;

  v8f acc = {};
  // A: lane=row i, halfs = u_lds[wbase + 16i + 32t + {g*8..g*8+7, g*8+16..g*8+23}]
  const int abase = wid * 256 + 16 * col + grp * 8;
  const int bbase = col * KREV_PITCH + 16 + grp * 8;
  #pragma unroll
  for (int t = 0; t < NSTEPS; ++t) {
    const _Float16* ap = &u_lds[abase + 32 * t];
    v16h a = cat16(*(const v8h*)(ap), *(const v8h*)(ap + 16));
    const _Float16* bp = &krevs[bbase + 32 * t];
    v16h bm = cat16(*(const v8h*)(bp), *(const v8h*)(bp + 16));
    acc = __builtin_amdgcn_wmma_f32_16x16x32_f16(false, a, false, bm,
                                                 (short)0, acc, false, false);
  }

  // Epilogue: + D[h]*u (fp32 source), exact GELU, store f16.
  const float dch = Dcoef[h];
  const int L0w = L0 + wid * 256;
  #pragma unroll
  for (int r = 0; r < 8; ++r) {
    const int n = L0w + 16 * (r + 8 * grp) + col;   // D: VGPR r = row r / r+8
    float y = acc[r] + dch * u[ubase + n];
    float g = 0.5f * y * (1.0f + erff(y * 0.70710678118654752f));
    G[ubase + n] = (_Float16)g;
  }
}

// ---------------------------------------------------------------------------
// Stage 3: out[b,o,l] = sum_f W[o,f] * G[b,f,l] + bias[o]   (fp32 out)
// 128x128 tile / workgroup, K=32 steps, double-buffered LDS, one barrier per
// step. W tile staged via async global->LDS when available; G tile is
// transposed while staging (l-major -> k-contiguous) so B frags are aligned
// ds_load_b128 pairs.
// ---------------------------------------------------------------------------
__global__ __launch_bounds__(256)
void gemm_kernel(const _Float16* __restrict__ G, const _Float16* __restrict__ W16,
                 const float* __restrict__ bias, float* __restrict__ out) {
  __shared__ _Float16 As[2][GM * KPITCH];   // [o_local][k] k-contiguous
  __shared__ _Float16 Bs[2][GN * KPITCH];   // [n_local][k] k-contiguous (transposed)

  const int tid = threadIdx.x;
  const int Nbase = blockIdx.x * GN;
  const int Mbase = blockIdx.y * GM;
  const int b = blockIdx.z;
  const int wid = tid >> 5, lane = tid & 31;
  const int col = lane & 15, grp = lane >> 4;
  const int waveM = wid & 3, waveN = wid >> 2;

  const int arow = tid & 127, ahp = tid >> 7;   // A staging role: 128 rows x 2
  const int bf = tid & 31,  bseg = tid >> 5;    // B staging role: 32 f x 8 segs

  v8f zero = {};
  v8f acc[2][4];
  #pragma unroll
  for (int mi = 0; mi < 2; ++mi)
    #pragma unroll
    for (int ni = 0; ni < 4; ++ni) acc[mi][ni] = zero;

  const _Float16* asrc = W16 + (size_t)(Mbase + arow) * H_DIM + ahp * 16;
  const _Float16* bsrc = G + (size_t)(b * H_DIM + bf) * L_DIM + Nbase + bseg * 16;
  _Float16* adst0 = &As[0][arow * KPITCH + ahp * 16];
  _Float16* adst1 = &As[1][arow * KPITCH + ahp * 16];

  v8h bg0, bg1;
#if !HAS_ASYNC_LDS
  v8h aw0, aw1;
#endif

  // ---- prologue: stage k-step 0 into buffer 0 ----
#if HAS_ASYNC_LDS
  async_copy16B(asrc, adst0);
#else
  aw0 = *(const v8h*)(asrc);
  aw1 = *(const v8h*)(asrc + 8);
  *(v8h*)(adst0)     = aw0;
  *(v8h*)(adst0 + 8) = aw1;
#endif
  bg0 = *(const v8h*)(bsrc);
  bg1 = *(const v8h*)(bsrc + 8);
  {
    _Float16* bdst = &Bs[0][bseg * 16 * KPITCH + bf];
    #pragma unroll
    for (int e = 0; e < 8; ++e) {
      bdst[e * KPITCH]       = bg0[e];
      bdst[(8 + e) * KPITCH] = bg1[e];
    }
  }
#if HAS_ASYNC_LDS
  __builtin_amdgcn_s_wait_asynccnt(0);
#endif
  __syncthreads();

  for (int kk = 0; kk < GKSTEPS; ++kk) {
    const int cur = kk & 1;
    const int nxt = cur ^ 1;
    const bool more = (kk + 1 < GKSTEPS);

    // issue next tile's global traffic early (overlaps the WMMAs below)
    if (more) {
      const _Float16* an = asrc + (kk + 1) * GK;
      const _Float16* bn = bsrc + (size_t)(kk + 1) * GK * L_DIM;
#if HAS_ASYNC_LDS
      async_copy16B(an, nxt ? adst1 : adst0);
#else
      aw0 = *(const v8h*)(an);
      aw1 = *(const v8h*)(an + 8);
#endif
      bg0 = *(const v8h*)(bn);
      bg1 = *(const v8h*)(bn + 8);
      if (kk + 2 < GKSTEPS) {   // gfx1250 global_prefetch path, 2 steps ahead
        __builtin_prefetch(asrc + (kk + 2) * GK, 0, 1);
        __builtin_prefetch(bsrc + (size_t)(kk + 2) * GK * L_DIM, 0, 1);
      }
    }

    // ---- compute from buffer `cur` ----
    v16h afrag[2], bfrag[4];
    #pragma unroll
    for (int mi = 0; mi < 2; ++mi) {
      const _Float16* ap = &As[cur][(32 * waveM + 16 * mi + col) * KPITCH + grp * 8];
      afrag[mi] = cat16(*(const v8h*)(ap), *(const v8h*)(ap + 16));
    }
    #pragma unroll
    for (int ni = 0; ni < 4; ++ni) {
      const _Float16* bp = &Bs[cur][(64 * waveN + 16 * ni + col) * KPITCH + grp * 8];
      bfrag[ni] = cat16(*(const v8h*)(bp), *(const v8h*)(bp + 16));
    }
    #pragma unroll
    for (int mi = 0; mi < 2; ++mi)
      #pragma unroll
      for (int ni = 0; ni < 4; ++ni)
        acc[mi][ni] = __builtin_amdgcn_wmma_f32_16x16x32_f16(
            false, afrag[mi], false, bfrag[ni], (short)0, acc[mi][ni], false, false);

    // ---- commit staged tile into buffer `nxt` ----
    if (more) {
#if !HAS_ASYNC_LDS
      _Float16* ad = nxt ? adst1 : adst0;
      *(v8h*)(ad)     = aw0;
      *(v8h*)(ad + 8) = aw1;
#endif
      _Float16* bdst = &Bs[nxt][bseg * 16 * KPITCH + bf];
      #pragma unroll
      for (int e = 0; e < 8; ++e) {
        bdst[e * KPITCH]       = bg0[e];
        bdst[(8 + e) * KPITCH] = bg1[e];
      }
#if HAS_ASYNC_LDS
      __builtin_amdgcn_s_wait_asynccnt(0);
#endif
    }
    __syncthreads();
  }

  // ---- epilogue: + bias, fp32 store ----
  #pragma unroll
  for (int mi = 0; mi < 2; ++mi) {
    #pragma unroll
    for (int r = 0; r < 8; ++r) {
      const int o = Mbase + 32 * waveM + 16 * mi + r + 8 * grp;
      const float bo = bias[o];
      #pragma unroll
      for (int ni = 0; ni < 4; ++ni) {
        const int l = Nbase + 64 * waveN + 16 * ni + col;
        out[(size_t)(b * H_DIM + o) * L_DIM + l] = acc[mi][ni][r] + bo;
      }
    }
  }
}

// ---------------------------------------------------------------------------
extern "C" void kernel_launch(void* const* d_in, const int* in_sizes, int n_in,
                              void* d_out, int out_size, void* d_ws, size_t ws_size,
                              hipStream_t stream) {
  const float* u   = (const float*)d_in[0];
  const float* k0  = (const float*)d_in[1];
  const float* k1  = (const float*)d_in[2];
  const float* k2  = (const float*)d_in[3];
  const float* k3  = (const float*)d_in[4];
  const float* Dc  = (const float*)d_in[5];
  const float* W   = (const float*)d_in[6];
  const float* bia = (const float*)d_in[7];
  float* out = (float*)d_out;

  // Workspace layout (all 16B aligned): krev | W16 | G   (~67 MB total)
  _Float16* KREV = (_Float16*)d_ws;                       // 1024*576
  _Float16* W16  = KREV + (size_t)H_DIM * KREV_PITCH;     // 1024*1024
  _Float16* G    = W16 + (size_t)H_DIM * H_DIM;           // 4*1024*8192

  build_filter_kernel<<<dim3(H_DIM), dim3(256), 0, stream>>>(k0, k1, k2, k3, KREV);
  convert_w_kernel<<<dim3((H_DIM * H_DIM + 255) / 256), dim3(256), 0, stream>>>(W, W16);
  conv_gelu_kernel<<<dim3(B_DIM * H_DIM, L_DIM / CONV_TILE), dim3(256), 0, stream>>>(
      u, Dc, KREV, G);
  gemm_kernel<<<dim3(L_DIM / GN, H_DIM / GM, B_DIM), dim3(256), 0, stream>>>(
      G, W16, bia, out);
}
